// GATLabelEncoder_35158602285584
// MI455X (gfx1250) — matrix operand order repaired and checked by reading, not compiled, not run on hardware
//
#include <hip/hip_runtime.h>
#include <hip/hip_bf16.h>

// ---------------------------------------------------------------------------
// 2-layer GAT for MI455X (gfx1250).  GEMMs via v_wmma_f32_16x16x32_bf16 with
// explicitly ping-ponged (statically indexed) double-buffered LDS staging:
// 1 barrier / K-step, static ds offsets, 4 WMMAs per loop body.
// Edge softmax/aggregation via L2 atomics (proj fits in 192MB L2).
// ---------------------------------------------------------------------------

typedef __attribute__((ext_vector_type(16))) __bf16 v16bf;
typedef __attribute__((ext_vector_type(8)))  float  v8f;

#define NNODES 50000
#define NEDGES 800000
#define HEADS  4
#define SLOPE  0.2f

#define MPAD   50048      // 782 * 64
#define K1P    320        // 300 -> pad 64  (10 K-steps, even)
#define F1     400
#define N1P    448        // 400 -> pad 64
#define DH1    100
#define K2P    448        // 400 -> pad 64  (14 K-steps, even)
#define F2     512
#define N2P    512
#define DH2    128

#define TILE_M 64
#define TILE_N 64
#define TILE_K 32

static __device__ __forceinline__ unsigned short f32_to_bf16(float f) {
    unsigned int u = __float_as_uint(f);
    unsigned int lsb = (u >> 16) & 1u;
    u += 0x7fffu + lsb;                    // round-to-nearest-even
    return (unsigned short)(u >> 16);
}
static __device__ __forceinline__ float lrelu(float v) {
    return v > 0.0f ? v : SLOPE * v;
}
// order-preserving float<->uint mapping for atomicMax on floats
static __device__ __forceinline__ unsigned int fmap(float f) {
    unsigned int u = __float_as_uint(f);
    return (u & 0x80000000u) ? ~u : (u | 0x80000000u);
}
static __device__ __forceinline__ float funmap(unsigned int u) {
    return __uint_as_float((u & 0x80000000u) ? (u & 0x7fffffffu) : ~u);
}

// --------------------------- utility kernels -------------------------------

__global__ void zero_u32_kernel(unsigned int* __restrict__ p, long long n) {
    long long i = (long long)blockIdx.x * blockDim.x + threadIdx.x;
    if (i < n) p[i] = 0u;
}

// f32 [rows x cols] -> bf16 zero-padded [rpad x cpad]
__global__ void pack_bf16_kernel(const float* __restrict__ src,
                                 unsigned short* __restrict__ dst,
                                 int rows, int cols, int rpad, int cpad) {
    long long i = (long long)blockIdx.x * blockDim.x + threadIdx.x;
    long long tot = (long long)rpad * cpad;
    if (i >= tot) return;
    int r = (int)(i / cpad), c = (int)(i % cpad);
    float v = (r < rows && c < cols) ? src[(long long)r * cols + c] : 0.0f;
    dst[i] = f32_to_bf16(v);
}

// f32 W[K x F] -> bf16 W^T zero-padded [npad x kpad]  (n-major)
__global__ void pack_bf16_t_kernel(const float* __restrict__ src,
                                   unsigned short* __restrict__ dst,
                                   int K, int F, int kpad, int npad) {
    long long i = (long long)blockIdx.x * blockDim.x + threadIdx.x;
    long long tot = (long long)npad * kpad;
    if (i >= tot) return;
    int n = (int)(i / kpad), k = (int)(i % kpad);
    float v = (k < K && n < F) ? src[(long long)k * F + n] : 0.0f;
    dst[i] = f32_to_bf16(v);
}

// ------------------------------ WMMA GEMM ----------------------------------
// C[M x ldc] = A[Mpad x K](bf16) * Bt[Npad x K](bf16, transposed weights);
// stores only n < Nvalid.  K must be a multiple of 2*TILE_K.
__global__ __launch_bounds__(256) void gemm_bf16_wmma_kernel(
    const unsigned short* __restrict__ A,
    const unsigned short* __restrict__ Bt,
    float* __restrict__ C,
    int M, int K, int Nvalid, int ldc)
{
    __shared__ unsigned short As0[TILE_M * TILE_K];  // ping
    __shared__ unsigned short Bs0[TILE_N * TILE_K];
    __shared__ unsigned short As1[TILE_M * TILE_K];  // pong
    __shared__ unsigned short Bs1[TILE_N * TILE_K];

    const int t    = threadIdx.x;
    const int wave = t >> 5;
    const int lane = t & 31;
    const int half = lane >> 4;
    const int r    = lane & 15;

    const int m0 = blockIdx.x * TILE_M;
    const int n0 = blockIdx.y * TILE_N;

    const int msub  = wave & 3;          // 0..3  -> 16-row subtile
    const int nsub0 = (wave >> 2) * 2;   // 0 or 2
    const int nsub1 = nsub0 + 1;

    const int row = t >> 2;              // 0..63 (staging row for both tiles)
    const int cg  = (t & 3) * 8;         // 0,8,16,24
    const int sidx = row * TILE_K + cg;  // staging LDS index

    // per-wave fragment base indices (static per buffer)
    const int aoff = (msub * 16 + r) * TILE_K + half * 16;  // +0 and +8
    const int boff0 = (nsub0 * 16 + r) * TILE_K + half * 16; // +0 and +8
    const int boff1 = (nsub1 * 16 + r) * TILE_K + half * 16;

    const unsigned short* pA = A  + (size_t)(m0 + row) * K + cg;
    const unsigned short* pB = Bt + (size_t)(n0 + row) * K + cg;

    // prologue: stage tile 0 into ping
    {
        uint4 av = *(const uint4*)(pA);
        uint4 bv = *(const uint4*)(pB);
        *(uint4*)(&As0[sidx]) = av;
        *(uint4*)(&Bs0[sidx]) = bv;
    }

    v8f acc0 = {}; v8f acc1 = {};
    union Frag { uint4 q[2]; v16bf v; } aF, bF0, bF1;

    for (int k0 = 0; k0 < K; k0 += 2 * TILE_K) {
        // ---- step A: compute from ping, prefetch k0+32 into pong ----
        {
            uint4 an = *(const uint4*)(pA + k0 + TILE_K);
            uint4 bn = *(const uint4*)(pB + k0 + TILE_K);
            __syncthreads();   // ping staged; pong reads (prev step B) retired
            aF.q[0]  = *(const uint4*)(&As0[aoff]);
            aF.q[1]  = *(const uint4*)(&As0[aoff + 8]);
            bF0.q[0] = *(const uint4*)(&Bs0[boff0]);
            bF0.q[1] = *(const uint4*)(&Bs0[boff0 + 8]);
            bF1.q[0] = *(const uint4*)(&Bs0[boff1]);
            bF1.q[1] = *(const uint4*)(&Bs0[boff1 + 8]);
            acc0 = __builtin_amdgcn_wmma_f32_16x16x32_bf16(
                       false, aF.v, false, bF0.v, (short)0, acc0, false, false);
            acc1 = __builtin_amdgcn_wmma_f32_16x16x32_bf16(
                       false, aF.v, false, bF1.v, (short)0, acc1, false, false);
            *(uint4*)(&As1[sidx]) = an;
            *(uint4*)(&Bs1[sidx]) = bn;
        }
        // ---- step B: compute from pong, prefetch k0+64 into ping ----
        {
            const bool more = (k0 + 2 * TILE_K) < K;
            uint4 an, bn;
            if (more) {
                an = *(const uint4*)(pA + k0 + 2 * TILE_K);
                bn = *(const uint4*)(pB + k0 + 2 * TILE_K);
            }
            __syncthreads();   // pong staged; ping reads (step A) retired
            aF.q[0]  = *(const uint4*)(&As1[aoff]);
            aF.q[1]  = *(const uint4*)(&As1[aoff + 8]);
            bF0.q[0] = *(const uint4*)(&Bs1[boff0]);
            bF0.q[1] = *(const uint4*)(&Bs1[boff0 + 8]);
            bF1.q[0] = *(const uint4*)(&Bs1[boff1]);
            bF1.q[1] = *(const uint4*)(&Bs1[boff1 + 8]);
            acc0 = __builtin_amdgcn_wmma_f32_16x16x32_bf16(
                       false, aF.v, false, bF0.v, (short)0, acc0, false, false);
            acc1 = __builtin_amdgcn_wmma_f32_16x16x32_bf16(
                       false, aF.v, false, bF1.v, (short)0, acc1, false, false);
            if (more) {
                *(uint4*)(&As0[sidx]) = an;
                *(uint4*)(&Bs0[sidx]) = bn;
            }
        }
    }

    // C/D layout: VGPR i -> M = half*8 + i, N = r
    #pragma unroll
    for (int i = 0; i < 8; ++i) {
        int orow = m0 + msub * 16 + half * 8 + i;
        if (orow < M) {
            int c0 = n0 + nsub0 * 16 + r;
            if (c0 < Nvalid) C[(size_t)orow * ldc + c0] = acc0[i];
            int c1 = n0 + nsub1 * 16 + r;
            if (c1 < Nvalid) C[(size_t)orow * ldc + c1] = acc1[i];
        }
    }
}

// ------------------------- attention edge kernels --------------------------

// el[n,h] = dot(proj[n,h,:], al[h,:]);  er likewise
__global__ void el_er_kernel(const float* __restrict__ proj,
                             const float* __restrict__ al,
                             const float* __restrict__ ar,
                             float* __restrict__ el, float* __restrict__ er,
                             int F, int Dh) {
    int i = blockIdx.x * blockDim.x + threadIdx.x;
    if (i >= NNODES * HEADS) return;
    int n = i / HEADS, h = i % HEADS;
    const float* p  = proj + (size_t)n * F + h * Dh;
    const float* pl = al + h * Dh;
    const float* pr = ar + h * Dh;
    float sl = 0.0f, sr = 0.0f;
    for (int d = 0; d < Dh; ++d) { float v = p[d]; sl += v * pl[d]; sr += v * pr[d]; }
    el[i] = sl; er[i] = sr;
}

// pass A: e = leaky(el[src]+er[dst]); store raw e; atomic segment-max (mapped)
__global__ void edge_max_kernel(const int* __restrict__ src, const int* __restrict__ dst,
                                const float* __restrict__ el, const float* __restrict__ er,
                                float* __restrict__ ebuf, unsigned int* __restrict__ mbuf) {
    long long i = (long long)blockIdx.x * blockDim.x + threadIdx.x;
    if (i >= (long long)NEDGES * HEADS) return;
    int e = (int)(i / HEADS), h = (int)(i % HEADS);
    int s = src[e], d = dst[e];
    float v = lrelu(el[s * HEADS + h] + er[d * HEADS + h]);
    ebuf[i] = v;
    atomicMax(&mbuf[d * HEADS + h], fmap(v));
}

// pass B: ex = exp(e - m[dst]); atomic segment-sum
__global__ void edge_expsum_kernel(const int* __restrict__ dst,
                                   float* __restrict__ ebuf,
                                   const unsigned int* __restrict__ mbuf,
                                   float* __restrict__ sbuf) {
    long long i = (long long)blockIdx.x * blockDim.x + threadIdx.x;
    if (i >= (long long)NEDGES * HEADS) return;
    int e = (int)(i / HEADS), h = (int)(i % HEADS);
    int d = dst[e];
    unsigned int mu = mbuf[d * HEADS + h];
    float mx = (mu == 0u) ? 0.0f : funmap(mu);   // no incoming edge -> 0 (isfinite fixup)
    float x = __expf(ebuf[i] - mx);
    ebuf[i] = x;
    atomicAdd(&sbuf[d * HEADS + h], x);
}

// pass C: agg[dst] += proj[src] * alpha   (one 128-thread block per edge)
__global__ __launch_bounds__(128) void edge_aggregate_kernel(
    const int* __restrict__ src, const int* __restrict__ dst,
    const float* __restrict__ ebuf, const float* __restrict__ sbuf,
    const float* __restrict__ proj, float* __restrict__ agg,
    int F, int Dh) {
    int e = blockIdx.x;
    int s = src[e], d = dst[e];
    __shared__ float a4[HEADS];
    if (threadIdx.x < HEADS)
        a4[threadIdx.x] = ebuf[(size_t)e * HEADS + threadIdx.x] /
                          (sbuf[d * HEADS + threadIdx.x] + 1e-9f);
    __syncthreads();
    for (int c = threadIdx.x; c < F; c += 128) {
        int h = c / Dh;
        atomicAdd(&agg[(size_t)d * F + c], proj[(size_t)s * F + c] * a4[h]);
    }
}

// out = leaky(agg + b)   (in place)
__global__ void finalize_lrelu_kernel(float* __restrict__ x,
                                      const float* __restrict__ b, int F) {
    long long i = (long long)blockIdx.x * blockDim.x + threadIdx.x;
    if (i >= (long long)NNODES * F) return;
    x[i] = lrelu(x[i] + b[(int)(i % F)]);
}
// out += b   (no activation, layer 2 output)
__global__ void finalize_bias_kernel(float* __restrict__ x,
                                     const float* __restrict__ b, int F) {
    long long i = (long long)blockIdx.x * blockDim.x + threadIdx.x;
    if (i >= (long long)NNODES * F) return;
    x[i] = x[i] + b[(int)(i % F)];
}

// ------------------------------- launcher ----------------------------------

static inline int blks(long long n, int b) { return (int)((n + b - 1) / b); }

extern "C" void kernel_launch(void* const* d_in, const int* in_sizes, int n_in,
                              void* d_out, int out_size, void* d_ws, size_t ws_size,
                              hipStream_t stream) {
    const float* init_emb = (const float*)d_in[0];
    const float* W1  = (const float*)d_in[1];
    const float* al1 = (const float*)d_in[2];
    const float* ar1 = (const float*)d_in[3];
    const float* b1  = (const float*)d_in[4];
    const float* W2  = (const float*)d_in[5];
    const float* al2 = (const float*)d_in[6];
    const float* ar2 = (const float*)d_in[7];
    const float* b2  = (const float*)d_in[8];
    const int*   src = (const int*)d_in[9];
    const int*   dst = (const int*)d_in[10];
    float* out = (float*)d_out;

    // workspace bump allocator (256B aligned)
    char* wsp = (char*)d_ws; size_t off = 0;
    auto alloc = [&](size_t bytes) -> void* {
        void* p = wsp + off; off += (bytes + 255) & ~(size_t)255; return p;
    };
    unsigned short* xb1  = (unsigned short*)alloc((size_t)MPAD * K1P * 2);
    unsigned short* xb2  = (unsigned short*)alloc((size_t)MPAD * K2P * 2);
    unsigned short* w1t  = (unsigned short*)alloc((size_t)N1P * K1P * 2);  // W1^T
    unsigned short* w2t  = (unsigned short*)alloc((size_t)N2P * K2P * 2);  // W2^T
    float* proj1 = (float*)alloc((size_t)NNODES * F1 * 4);
    float* proj2 = (float*)alloc((size_t)NNODES * F2 * 4);
    float* agg1  = (float*)alloc((size_t)NNODES * F1 * 4);  // becomes layer-1 output
    float* el    = (float*)alloc((size_t)NNODES * HEADS * 4);
    float* er    = (float*)alloc((size_t)NNODES * HEADS * 4);
    unsigned int* mbuf = (unsigned int*)alloc((size_t)NNODES * HEADS * 4);
    float* sbuf  = (float*)alloc((size_t)NNODES * HEADS * 4);
    float* ebuf  = (float*)alloc((size_t)NEDGES * HEADS * 4);
    (void)ws_size; (void)in_sizes; (void)n_in; (void)out_size;

    const int NH = NNODES * HEADS;
    const long long EH = (long long)NEDGES * HEADS;

    // ---------------- layer 1 ----------------
    pack_bf16_t_kernel<<<blks((long long)N1P * K1P, 256), 256, 0, stream>>>(W1, w1t, 300, F1, K1P, N1P);
    pack_bf16_kernel<<<blks((long long)MPAD * K1P, 256), 256, 0, stream>>>(init_emb, xb1, NNODES, 300, MPAD, K1P);

    gemm_bf16_wmma_kernel<<<dim3(MPAD / TILE_M, N1P / TILE_N), 256, 0, stream>>>(
        xb1, w1t, proj1, NNODES, K1P, F1, F1);

    el_er_kernel<<<blks(NH, 256), 256, 0, stream>>>(proj1, al1, ar1, el, er, F1, DH1);

    zero_u32_kernel<<<blks(NH, 256), 256, 0, stream>>>(mbuf, NH);
    zero_u32_kernel<<<blks(NH, 256), 256, 0, stream>>>((unsigned int*)sbuf, NH);
    zero_u32_kernel<<<blks((long long)NNODES * F1, 256), 256, 0, stream>>>((unsigned int*)agg1, (long long)NNODES * F1);

    edge_max_kernel<<<blks(EH, 256), 256, 0, stream>>>(src, dst, el, er, ebuf, mbuf);
    edge_expsum_kernel<<<blks(EH, 256), 256, 0, stream>>>(dst, ebuf, mbuf, sbuf);
    edge_aggregate_kernel<<<NEDGES, 128, 0, stream>>>(src, dst, ebuf, sbuf, proj1, agg1, F1, DH1);

    finalize_lrelu_kernel<<<blks((long long)NNODES * F1, 256), 256, 0, stream>>>(agg1, b1, F1);

    // ---------------- layer 2 ----------------
    pack_bf16_t_kernel<<<blks((long long)N2P * K2P, 256), 256, 0, stream>>>(W2, w2t, F1, F2, K2P, N2P);
    pack_bf16_kernel<<<blks((long long)MPAD * K2P, 256), 256, 0, stream>>>(agg1, xb2, NNODES, F1, MPAD, K2P);

    gemm_bf16_wmma_kernel<<<dim3(MPAD / TILE_M, N2P / TILE_N), 256, 0, stream>>>(
        xb2, w2t, proj2, NNODES, K2P, F2, F2);

    el_er_kernel<<<blks(NH, 256), 256, 0, stream>>>(proj2, al2, ar2, el, er, F2, DH2);

    zero_u32_kernel<<<blks(NH, 256), 256, 0, stream>>>(mbuf, NH);
    zero_u32_kernel<<<blks(NH, 256), 256, 0, stream>>>((unsigned int*)sbuf, NH);
    zero_u32_kernel<<<blks((long long)NNODES * F2, 256), 256, 0, stream>>>((unsigned int*)out, (long long)NNODES * F2);

    edge_max_kernel<<<blks(EH, 256), 256, 0, stream>>>(src, dst, el, er, ebuf, mbuf);
    edge_expsum_kernel<<<blks(EH, 256), 256, 0, stream>>>(dst, ebuf, mbuf, sbuf);
    edge_aggregate_kernel<<<NEDGES, 128, 0, stream>>>(src, dst, ebuf, sbuf, proj2, out, F2, DH2);

    finalize_bias_kernel<<<blks((long long)NNODES * F2, 256), 256, 0, stream>>>(out, b2, F2);
}
